// Conv2d_11450382811540
// MI455X (gfx1250) — matrix operand order repaired and checked by reading, not compiled, not run on hardware
//
#include <hip/hip_runtime.h>

// ---------------------------------------------------------------------------
// Implicit-GEMM 3x3 conv for MI455X (gfx1250, wave32, WMMA).
//
// GEMM view:  out[m, n] = sum_r  A[m, r] * B[r, n]
//   m = (b, oh, ow)   (M tile dim = ow, contiguous in memory)
//   n = cout (256)
//   r = cin*9 + kh*3 + kw  (1152) -- the reference's flat-buffer weight order.
//
// Precision: fp32 emulated via bf16 split (hi = truncate-to-bf16, lo =
// truncate(f - hi); residual exact in f32). 3 WMMAs per K-chunk per C-tile:
//   ah*bh + ah*bl + al*bh        (dropped al*bl term ~2^-16 relative)
// Uses V_WMMA_F32_16X16X32_BF16, the fast K=32 tensor path.
//
// Blocking: one wave = 32(ow) x 128(cout) = 2 M-tiles x 8 N-tiles.
//   - each B fragment pair feeds 6 WMMAs (both M-tiles)
//   - each A fragment build feeds 24 WMMAs (8 N-tiles)
//   - 36 K-chunks x 48 WMMAs = 1728 WMMAs per wave.
// ---------------------------------------------------------------------------

typedef __attribute__((ext_vector_type(16))) __bf16   v16bf;
typedef __attribute__((ext_vector_type(8)))  float    v8f;
typedef __attribute__((ext_vector_type(8)))  unsigned v8u;

#define CIN   128
#define COUT  256
#define HW    128
#define VALID 125  // reference writes only [0,125)^2; border must be 0

static __device__ __forceinline__ __bf16 f32_to_bf16_rne(float f) {
  unsigned u = __builtin_bit_cast(unsigned, f);
  unsigned r = (u + 0x7FFFu + ((u >> 16) & 1u)) >> 16;
  unsigned short s = (unsigned short)r;
  return __builtin_bit_cast(__bf16, s);
}
static __device__ __forceinline__ float bf16_to_f32(__bf16 h) {
  unsigned short s = __builtin_bit_cast(unsigned short, h);
  unsigned u = ((unsigned)s) << 16;
  return __builtin_bit_cast(float, u);
}

// ---------------------------------------------------------------------------
// Weight prepack: (1152,256) f32 -> hi/lo bf16 B-fragments in WMMA register
// order. Fragment fb = chunk*16 + ntile, chunk = (kh*3+kw)*4 + cin0/32.
// Per fragment: 32 lanes x 16 bf16 (hi), lo fragment at +512 elements.
// B lane layout (32x16):  N=lane&15, K=8h+j (j<8), 16+8h+(j-8) (j>=8), h=lane/16.
// Total 1.13 MB in d_ws.
// ---------------------------------------------------------------------------
__global__ void conv_prepack_w(const float* __restrict__ Kw,
                               __bf16* __restrict__ wpk) {
  int t    = blockIdx.x * 256 + threadIdx.x;   // 36*16*32*16 = 294912
  int j    = t & 15;
  int l    = (t >> 4) & 31;
  int nt   = (t >> 9) & 15;
  int ck   = t >> 13;                          // 0..35
  int c32  = ck & 3;
  int kpos = ck >> 2;                          // kh*3 + kw
  int kidx = ((j < 8) ? j : (j + 8)) + ((l >= 16) ? 8 : 0);
  int cin  = (c32 << 5) + kidx;
  int r    = cin * 9 + kpos;
  int c    = (nt << 4) + (l & 15);
  float v  = Kw[r * COUT + c];
  __bf16 hi = f32_to_bf16_rne(v);
  __bf16 lo = f32_to_bf16_rne(v - bf16_to_f32(hi));
  size_t base = ((size_t)(ck * 16 + nt)) * 1024 + (size_t)l * 16 + j;
  wpk[base]       = hi;
  wpk[base + 512] = lo;
}

// Build one A fragment pair (hi, lo) for 16 ow positions at row pointer xp.
// xp must be a legal address (clamped); `ok` masks padded positions to 0.
static __device__ __forceinline__ void build_a_frag(
    const float* __restrict__ p, bool ok, v16bf& ahi, v16bf& alo) {
  float f[16];
  #pragma unroll
  for (int j = 0; j < 16; ++j) {
    const int kk = (j < 8) ? j : (j + 8);        // ISA A lane layout
    float v = p[(size_t)kk << 14];               // imm-offset loads
    f[j] = ok ? v : 0.f;                         // v_cndmask, no branch
  }
  v8u hiu, lou;
  #pragma unroll
  for (int q = 0; q < 8; ++q) {
    unsigned u0 = __builtin_bit_cast(unsigned, f[2 * q]);
    unsigned u1 = __builtin_bit_cast(unsigned, f[2 * q + 1]);
    float h0 = __builtin_bit_cast(float, u0 & 0xFFFF0000u);
    float h1 = __builtin_bit_cast(float, u1 & 0xFFFF0000u);
    unsigned r0 = __builtin_bit_cast(unsigned, f[2 * q] - h0);
    unsigned r1 = __builtin_bit_cast(unsigned, f[2 * q + 1] - h1);
    hiu[q] = __builtin_amdgcn_perm(u1, u0, 0x07060302u);   // top16 pair
    lou[q] = __builtin_amdgcn_perm(r1, r0, 0x07060302u);
  }
  ahi = __builtin_bit_cast(v16bf, hiu);
  alo = __builtin_bit_cast(v16bf, lou);
}

// ---------------------------------------------------------------------------
// Main kernel: 32(ow) x 128(cout) per wave, 8 waves/block.
// ---------------------------------------------------------------------------
__global__ void __launch_bounds__(256)
conv_wmma(const float* __restrict__ x,
          const __bf16* __restrict__ wpk,
          float* __restrict__ out) {
  const int lane  = threadIdx.x & 31;
  const int wave  = threadIdx.x >> 5;
  const int tile  = blockIdx.x * 8 + wave;     // 16384 wave-tiles total
  const int ngrp  = tile & 1;                  // couts [0,128) or [128,256)
  const int mtile = tile >> 1;                 // 8192 spatial supertiles
  const int mbase = mtile << 5;                // m = b*16384 + oh*128 + ow
  const int b     = mbase >> 14;
  const int oh    = (mbase >> 7) & 127;
  const int ow0   = mbase & 127;               // multiple of 32
  const int khalf = lane >> 4;                 // which K-half this lane holds
  const int ml    = lane & 15;                 // A row == ow offset
  const int nbase = ngrp << 3;                 // first of 8 ntiles

  // Fold b and the lane's K-half (khalf*8 cin) into the base pointer.
  const float* xb = x + ((size_t)b << 21) + ((size_t)khalf << 17);

  v8f acc[2][8];
  #pragma unroll
  for (int mt = 0; mt < 2; ++mt)
    #pragma unroll
    for (int t = 0; t < 8; ++t) acc[mt][t] = (v8f){};

  #pragma unroll
  for (int kh = 0; kh < 3; ++kh) {
    const int  ih    = oh + kh - 1;
    const bool rowok = (unsigned)ih < (unsigned)HW;
    const int  ihc   = rowok ? ih : 0;          // clamped (wave-uniform)
    const float* xrow = xb + ihc * HW;
    #pragma unroll
    for (int kw = 0; kw < 3; ++kw) {
      // per-M-tile column coordinates / masks / clamped row pointers
      const int  iw0  = ow0 + ml + kw - 1;
      const int  iw1  = iw0 + 16;
      const bool ok0  = rowok && ((unsigned)iw0 < (unsigned)HW);
      const bool ok1  = rowok && ((unsigned)iw1 < (unsigned)HW);
      const float* xp0 = xrow + (((unsigned)iw0 < (unsigned)HW) ? iw0 : 0);
      const float* xp1 = xrow + (((unsigned)iw1 < (unsigned)HW) ? iw1 : 0);
      const int  kpos = kh * 3 + kw;

      #pragma unroll
      for (int c32 = 0; c32 < 4; ++c32) {
        const size_t coff = ((size_t)c32 << 19);   // + c0*16384
        v16bf ahi0, alo0, ahi1, alo1;
        build_a_frag(xp0 + coff, ok0, ahi0, alo0);
        build_a_frag(xp1 + coff, ok1, ahi1, alo1);

        // ---- 8 N-tiles x 2 M-tiles reuse fragments: 48 WMMAs per chunk ----
        const int fb0 = ((kpos << 2) + c32) * 16 + nbase;
        #pragma unroll
        for (int t = 0; t < 8; ++t) {
          const __bf16* bp = wpk + ((size_t)(fb0 + t) << 10)
                                 + ((size_t)lane << 4);
          v16bf bhi = *(const v16bf*)(bp);
          v16bf blo = *(const v16bf*)(bp + 512);
          acc[0][t] = __builtin_amdgcn_wmma_f32_16x16x32_bf16(
                        false, ahi0, false, bhi, (short)0, acc[0][t], false, false);
          acc[1][t] = __builtin_amdgcn_wmma_f32_16x16x32_bf16(
                        false, ahi1, false, bhi, (short)0, acc[1][t], false, false);
          acc[0][t] = __builtin_amdgcn_wmma_f32_16x16x32_bf16(
                        false, ahi0, false, blo, (short)0, acc[0][t], false, false);
          acc[1][t] = __builtin_amdgcn_wmma_f32_16x16x32_bf16(
                        false, ahi1, false, blo, (short)0, acc[1][t], false, false);
          acc[0][t] = __builtin_amdgcn_wmma_f32_16x16x32_bf16(
                        false, alo0, false, bhi, (short)0, acc[0][t], false, false);
          acc[1][t] = __builtin_amdgcn_wmma_f32_16x16x32_bf16(
                        false, alo1, false, bhi, (short)0, acc[1][t], false, false);
        }
      }
    }
  }

  // ---- Store: C/D layout N = lane&15, rows M = r + 8*khalf = ow offset.
  // Each lane's 8 accs per tile are consecutive ow -> two b128 stores.
  // Masked border stores also clear the harness's 0xAA poison to 0.0.
  const bool rowok = (oh < VALID);
  #pragma unroll
  for (int mt = 0; mt < 2; ++mt) {
    const int owb = ow0 + (mt << 4) + (khalf << 3);
    const size_t outrow = ((size_t)oh << 7) + owb;
    #pragma unroll
    for (int t = 0; t < 8; ++t) {
      const int n = ((nbase + t) << 4) + ml;
      float* op = out + (((size_t)(b * COUT + n)) << 14) + outrow;
      float4 s0, s1;
      s0.x = (rowok && owb + 0 < VALID) ? acc[mt][t][0] : 0.f;
      s0.y = (rowok && owb + 1 < VALID) ? acc[mt][t][1] : 0.f;
      s0.z = (rowok && owb + 2 < VALID) ? acc[mt][t][2] : 0.f;
      s0.w = (rowok && owb + 3 < VALID) ? acc[mt][t][3] : 0.f;
      s1.x = (rowok && owb + 4 < VALID) ? acc[mt][t][4] : 0.f;
      s1.y = (rowok && owb + 5 < VALID) ? acc[mt][t][5] : 0.f;
      s1.z = (rowok && owb + 6 < VALID) ? acc[mt][t][6] : 0.f;
      s1.w = (rowok && owb + 7 < VALID) ? acc[mt][t][7] : 0.f;
      *(float4*)(op)     = s0;
      *(float4*)(op + 4) = s1;
    }
  }
}

extern "C" void kernel_launch(void* const* d_in, const int* in_sizes, int n_in,
                              void* d_out, int out_size, void* d_ws, size_t ws_size,
                              hipStream_t stream) {
  const float* x  = (const float*)d_in[0];   // (16,128,128,128) f32
  const float* Kw = (const float*)d_in[1];   // flat (1152,256) f32
  float* out      = (float*)d_out;           // (16,256,128,128) f32
  __bf16* wpk     = (__bf16*)d_ws;           // 1.13 MB prepacked weights

  conv_prepack_w<<<1152, 256, 0, stream>>>(Kw, wpk);

  // 8192 M-supertiles x 2 cout-groups = 16384 waves / 8 waves per block
  conv_wmma<<<2048, 256, 0, stream>>>(x, wpk, out);
}
